// MultiHeadAttention_26036091749073
// MI455X (gfx1250) — matrix operand order repaired and checked by reading, not compiled, not run on hardware
//
#include <hip/hip_runtime.h>
#include <math.h>

// ---------------------------------------------------------------------------
// MHA with RoPE + causal flash attention for MI455X (gfx1250, wave32, WMMA).
//   B=2, T=2048, C=1024, H=16, Dh=64
// All GEMMs: v_wmma_f32_16x16x32_bf16 (bf16 in, fp32 accumulate), software-
// pipelined so loads overlap matrix ops. Transposes are LDS-tiled.
// ---------------------------------------------------------------------------

#define BATCH   2
#define SEQ     2048
#define DMODEL  1024
#define NHEADS  16
#define HDIM    64
#define ROWS    (BATCH * SEQ)          // 4096
#define QKVN    (3 * DMODEL)           // 3072

typedef __bf16  bf16;
typedef __attribute__((ext_vector_type(16))) __bf16 v16bf;
typedef __attribute__((ext_vector_type(4)))  __bf16 bf16x4;
typedef __attribute__((ext_vector_type(8)))  float  v8f;
typedef __attribute__((ext_vector_type(4)))  float  f32x4;

// 32-byte WMMA operand fragment (16 bf16) loadable as two 16-byte chunks.
union Frag {
    v16bf v;
    f32x4 f[2];
};

static __device__ __forceinline__ v8f wmma_bf16(v16bf a, v16bf b, v8f c) {
    return __builtin_amdgcn_wmma_f32_16x16x32_bf16(
        false, a, false, b, (short)0, c, false, false);
}

// ---------------------------------------------------------------------------
// fp32 -> bf16 conversion, 4 elements/thread (b128 in, b64 out)
// ---------------------------------------------------------------------------
__global__ void k_cvt_bf16_v4(const float* __restrict__ in, bf16* __restrict__ out,
                              int n4) {
    int i = blockIdx.x * blockDim.x + threadIdx.x;
    if (i >= n4) return;
    f32x4 v = ((const f32x4*)in)[i];
    bf16x4 o;
    o[0] = (bf16)v[0]; o[1] = (bf16)v[1]; o[2] = (bf16)v[2]; o[3] = (bf16)v[3];
    ((bf16x4*)out)[i] = o;
}

// ---------------------------------------------------------------------------
// LDS-tiled transpose+convert: out[n*K + k] = (bf16) in[k*N + n]
// 32x32 tiles, coalesced fp32 reads and bf16 writes.
// ---------------------------------------------------------------------------
__global__ __launch_bounds__(256)
void k_transpose_bf16(const float* __restrict__ in, bf16* __restrict__ out,
                      int K, int N) {
    __shared__ bf16 tile[32][33];
    const int tilesN = N >> 5;
    const int bk = blockIdx.x / tilesN;
    const int bn = blockIdx.x - bk * tilesN;
    const int c  = threadIdx.x & 31;
    const int r0 = threadIdx.x >> 5;           // 0..7
#pragma unroll
    for (int rr = 0; rr < 32; rr += 8) {
        const int r = r0 + rr;
        tile[r][c] = (bf16)in[(size_t)(bk * 32 + r) * N + bn * 32 + c];
    }
    __syncthreads();
#pragma unroll
    for (int rr = 0; rr < 32; rr += 8) {
        const int r = r0 + rr;
        out[(size_t)(bn * 32 + r) * K + bk * 32 + c] = tile[c][r];
    }
}

// ---------------------------------------------------------------------------
// V transpose: qkv fp32 [ROWS][3072] V-section -> vbt [bh][64][T] bf16.
// LDS-tiled so both sides are coalesced.
// ---------------------------------------------------------------------------
__global__ __launch_bounds__(256)
void k_v_transpose(const float* __restrict__ qkv, bf16* __restrict__ vbt) {
    __shared__ bf16 tile[32][33];
    const int bid = blockIdx.x;            // 32 bh * 2 d-tiles * 64 t-tiles
    const int tt  = bid & 63;
    const int dt  = (bid >> 6) & 1;
    const int bh  = bid >> 7;
    const int b   = bh >> 4;
    const int h   = bh & 15;
    const int c   = threadIdx.x & 31;
    const int r0  = threadIdx.x >> 5;
#pragma unroll
    for (int rr = 0; rr < 32; rr += 8) {
        const int r = r0 + rr;            // t_local
        tile[r][c] = (bf16)qkv[(size_t)(b * SEQ + tt * 32 + r) * QKVN
                               + 2 * DMODEL + h * HDIM + dt * 32 + c];
    }
    __syncthreads();
#pragma unroll
    for (int rr = 0; rr < 32; rr += 8) {
        const int r = r0 + rr;            // d_local
        vbt[((size_t)bh * HDIM + dt * 32 + r) * SEQ + tt * 32 + c] = tile[c][r];
    }
}

// ---------------------------------------------------------------------------
// GEMM: C[M][N] = A[M][K] @ Bt[N][K]^T + bias[N]
// Wave tile 32x64 (8 accumulators). K-loop processes 64 per iteration with
// ping-pong fragment buffers: loads for k+32 are in flight during WMMAs on k.
// Requires K % 64 == 0, M % 32 == 0, N % 64 == 0.
// ---------------------------------------------------------------------------
__global__ __launch_bounds__(128)
void k_gemm_bf16(const bf16* __restrict__ A, const bf16* __restrict__ Bt,
                 const float* __restrict__ bias, float* __restrict__ C,
                 int M, int N, int K) {
    const int lane = threadIdx.x & 31;
    const int wv   = threadIdx.x >> 5;
    const int wg   = blockIdx.x * 4 + wv;
    const int tilesN = N >> 6;
    const int tm = wg / tilesN;
    const int tn = wg - tm * tilesN;
    if (tm * 32 >= M) return;

    const int mloc = lane & 15;
    const int hi   = lane >> 4;

    const bf16* arow0 = A + (size_t)(tm * 32 + mloc) * K;
    const bf16* arow1 = A + (size_t)(tm * 32 + 16 + mloc) * K;
    const bf16* bbase = Bt + (size_t)(tn * 64 + mloc) * K;

    v8f acc[8];
#pragma unroll
    for (int i = 0; i < 8; ++i) acc[i] = (v8f){};

    Frag a0[2], a1[2], b0[4], b1[4];

    auto loadA = [&](Frag* a, int k0) {
        a[0].f[0] = *(const f32x4*)(arow0 + k0 + hi * 16);
        a[0].f[1] = *(const f32x4*)(arow0 + k0 + hi * 16 + 8);
        a[1].f[0] = *(const f32x4*)(arow1 + k0 + hi * 16);
        a[1].f[1] = *(const f32x4*)(arow1 + k0 + hi * 16 + 8);
    };
    auto loadB = [&](Frag* bb, int k0) {
#pragma unroll
        for (int nt = 0; nt < 4; ++nt) {
            const bf16* p = bbase + (size_t)nt * 16 * K + k0 + hi * 16;
            bb[nt].f[0] = *(const f32x4*)(p);
            bb[nt].f[1] = *(const f32x4*)(p + 8);
        }
    };
    auto mma = [&](Frag* a, Frag* bb) {
#pragma unroll
        for (int nt = 0; nt < 4; ++nt) {
            acc[nt]     = wmma_bf16(a[0].v, bb[nt].v, acc[nt]);
            acc[4 + nt] = wmma_bf16(a[1].v, bb[nt].v, acc[4 + nt]);
        }
    };

    loadA(a0, 0);
    loadB(b0, 0);
    for (int k0 = 0; k0 + 64 <= K; k0 += 64) {
        loadA(a1, k0 + 32);
        loadB(b1, k0 + 32);
        mma(a0, b0);                       // waits only on set0; set1 in flight
        if (k0 + 64 < K) { loadA(a0, k0 + 64); loadB(b0, k0 + 64); }
        mma(a1, b1);
    }

#pragma unroll
    for (int mb = 0; mb < 2; ++mb) {
#pragma unroll
        for (int nt = 0; nt < 4; ++nt) {
            const int col = tn * 64 + nt * 16 + mloc;
            const float bv = bias[col];
#pragma unroll
            for (int r = 0; r < 8; ++r) {
                const int row = tm * 32 + mb * 16 + r + hi * 8;
                C[(size_t)row * N + col] = acc[mb * 4 + nt][r] + bv;
            }
        }
    }
}

// ---------------------------------------------------------------------------
// RoPE: qkv fp32 -> qb [bh][T][64] (scaled 1/8), kb [bh][T][64], both bf16.
// (V handled by k_v_transpose.)
// ---------------------------------------------------------------------------
__global__ void k_rope_qk(const float* __restrict__ qkv,
                          bf16* __restrict__ qb, bf16* __restrict__ kb) {
    const int i = blockIdx.x * blockDim.x + threadIdx.x;
    if (i >= BATCH * NHEADS * SEQ * HDIM) return;
    const int d  = i & 63;
    const int t  = (i >> 6) & (SEQ - 1);
    const int bh = i >> 17;
    const int b  = bh >> 4;
    const int h  = bh & 15;

    const size_t row = (size_t)(b * SEQ + t) * QKVN;
    const int cbase  = h * HDIM;
    const int d2     = (d < 32) ? d + 32 : d - 32;

    const float xq  = qkv[row + cbase + d];
    const float xq2 = qkv[row + cbase + d2];
    const float xk  = qkv[row + DMODEL + cbase + d];
    const float xk2 = qkv[row + DMODEL + cbase + d2];

    const int j = d & 31;
    const float theta = __powf(10000.f, -(float)(2 * j) * (1.0f / 64.0f));
    float s, c;
    __sincosf((float)t * theta, &s, &c);
    const float rq = (d < 32) ? -xq2 : xq2;
    const float rk = (d < 32) ? -xk2 : xk2;

    const size_t qi = ((size_t)bh * SEQ + t) * HDIM + d;
    qb[qi] = (bf16)((xq * c + rq * s) * 0.125f);   // fold 1/sqrt(64) into Q
    kb[qi] = (bf16)(xk * c + rk * s);
}

// ---------------------------------------------------------------------------
// Causal flash attention. One wave per 16-query tile of one (b,h).
// Pipelined 32-key chunks: V frags load before S WMMAs, next-chunk K frags
// load right after them; softmax VALU covers the latency. P re-laid-out
// D->A via per-wave LDS (in-order within a wave).
// ---------------------------------------------------------------------------
__global__ __launch_bounds__(128)
void k_flash_attn(const bf16* __restrict__ qb, const bf16* __restrict__ kb,
                  const bf16* __restrict__ vbt, bf16* __restrict__ attn) {
    __shared__ __align__(16) bf16 plds[4][16][40];

    const int lane = threadIdx.x & 31;
    const int wv   = threadIdx.x >> 5;
    const int wg   = blockIdx.x * 4 + wv;
    const int bh   = wg >> 7;
    const int qt   = wg & 127;
    const int qrow = qt << 4;
    const int b    = bh >> 4;
    const int h    = bh & 15;
    const int n    = lane & 15;
    const int hi   = lane >> 4;

    const bf16* qr = qb + ((size_t)bh * SEQ + qrow + n) * HDIM;
    Frag aq0, aq1;
    aq0.f[0] = *(const f32x4*)(qr + hi * 16);
    aq0.f[1] = *(const f32x4*)(qr + hi * 16 + 8);
    aq1.f[0] = *(const f32x4*)(qr + 32 + hi * 16);
    aq1.f[1] = *(const f32x4*)(qr + 32 + hi * 16 + 8);

    v8f o[4];
    o[0] = (v8f){}; o[1] = (v8f){}; o[2] = (v8f){}; o[3] = (v8f){};
    float mrow[8], lrow[8];
#pragma unroll
    for (int r = 0; r < 8; ++r) { mrow[r] = -1e30f; lrow[r] = 0.f; }

    const bf16* kbase = kb  + (size_t)bh * SEQ * HDIM;
    const bf16* vbase = vbt + (size_t)bh * HDIM * SEQ;

    Frag kc[4], kn[4];   // current / next chunk K fragments
    auto loadK = [&](Frag* kf, int kst) {
        const bf16* kr0 = kbase + (size_t)(kst + n) * HDIM;
        const bf16* kr1 = kbase + (size_t)(kst + 16 + n) * HDIM;
        kf[0].f[0] = *(const f32x4*)(kr0 + hi * 16);
        kf[0].f[1] = *(const f32x4*)(kr0 + hi * 16 + 8);
        kf[1].f[0] = *(const f32x4*)(kr0 + 32 + hi * 16);
        kf[1].f[1] = *(const f32x4*)(kr0 + 32 + hi * 16 + 8);
        kf[2].f[0] = *(const f32x4*)(kr1 + hi * 16);
        kf[2].f[1] = *(const f32x4*)(kr1 + hi * 16 + 8);
        kf[3].f[0] = *(const f32x4*)(kr1 + 32 + hi * 16);
        kf[3].f[1] = *(const f32x4*)(kr1 + 32 + hi * 16 + 8);
    };

    const int nch = (qrow + 16 + 31) >> 5;
    loadK(kc, 0);

    for (int c = 0; c < nch; ++c) {
        const int kst = c * 32;

        // ---- early V fragment loads (consumed only after softmax)
        Frag bv[4];
#pragma unroll
        for (int nt = 0; nt < 4; ++nt) {
            const bf16* vr = vbase + (size_t)(nt * 16 + n) * SEQ + kst + hi * 16;
            bv[nt].f[0] = *(const f32x4*)(vr);
            bv[nt].f[1] = *(const f32x4*)(vr + 8);
        }

        // ---- S tiles (waits only on kc; bv + kn stay in flight)
        v8f s0 = (v8f){}, s1 = (v8f){};
        s0 = wmma_bf16(aq0.v, kc[0].v, s0);
        s0 = wmma_bf16(aq1.v, kc[1].v, s0);
        s1 = wmma_bf16(aq0.v, kc[2].v, s1);
        s1 = wmma_bf16(aq1.v, kc[3].v, s1);

        // ---- issue next-chunk K loads + L2 prefetch two chunks ahead
        if (c + 1 < nch) loadK(kn, kst + 32);
        if (c + 2 < nch) {
            __builtin_prefetch(kbase + (size_t)(kst + 64 + n * 2) * HDIM, 0, 3);
            __builtin_prefetch(vbase + (size_t)(n * 16 + hi * 8) * SEQ + kst + 64, 0, 3);
        }

        // ---- causal mask (uniform branch; only diagonal-crossing chunks)
        if (kst + 31 > qrow) {
#pragma unroll
            for (int r = 0; r < 8; ++r) {
                const int trow = qrow + r + hi * 8;
                s0[r] = (kst + n      <= trow) ? s0[r] : -1e30f;
                s1[r] = (kst + 16 + n <= trow) ? s1[r] : -1e30f;
            }
        }

        // ---- online softmax update (row stats via 16-lane shfl reductions)
#pragma unroll
        for (int r = 0; r < 8; ++r) {
            float mx = fmaxf(s0[r], s1[r]);
            mx = fmaxf(mx, __shfl_xor(mx, 1));
            mx = fmaxf(mx, __shfl_xor(mx, 2));
            mx = fmaxf(mx, __shfl_xor(mx, 4));
            mx = fmaxf(mx, __shfl_xor(mx, 8));
            const float mnew  = fmaxf(mrow[r], mx);
            const float alpha = __expf(mrow[r] - mnew);
            mrow[r] = mnew;
            const float p0 = __expf(s0[r] - mnew);
            const float p1 = __expf(s1[r] - mnew);
            float rs = p0 + p1;
            rs += __shfl_xor(rs, 1);
            rs += __shfl_xor(rs, 2);
            rs += __shfl_xor(rs, 4);
            rs += __shfl_xor(rs, 8);
            lrow[r] = lrow[r] * alpha + rs;
            o[0][r] *= alpha; o[1][r] *= alpha; o[2][r] *= alpha; o[3][r] *= alpha;
            plds[wv][r + hi * 8][n]      = (bf16)p0;
            plds[wv][r + hi * 8][16 + n] = (bf16)p1;
        }

        __builtin_amdgcn_wave_barrier();   // LDS ops are in-order within a wave
        Frag ap;
        const bf16* pr = &plds[wv][n][0];
        ap.f[0] = *(const f32x4*)(pr + hi * 16);
        ap.f[1] = *(const f32x4*)(pr + hi * 16 + 8);
        __builtin_amdgcn_wave_barrier();

        // ---- O += P @ V  (bv loads issued ~200 VALU ops ago)
#pragma unroll
        for (int nt = 0; nt < 4; ++nt)
            o[nt] = wmma_bf16(ap.v, bv[nt].v, o[nt]);

        // ---- rotate K buffers
        if (c + 1 < nch) {
#pragma unroll
            for (int i = 0; i < 4; ++i) kc[i].v = kn[i].v;
        }
    }

    // ---- epilogue: normalize, write bf16 [B*T][C]
#pragma unroll
    for (int r = 0; r < 8; ++r) {
        const int trow  = qrow + r + hi * 8;
        const float inv = 1.0f / lrow[r];
        const size_t orow = (size_t)(b * SEQ + trow) * DMODEL + h * HDIM;
#pragma unroll
        for (int nt = 0; nt < 4; ++nt)
            attn[orow + nt * 16 + n] = (bf16)(o[nt][r] * inv);
    }
}

// ---------------------------------------------------------------------------
// Host launcher
// ---------------------------------------------------------------------------
extern "C" void kernel_launch(void* const* d_in, const int* in_sizes, int n_in,
                              void* d_out, int out_size, void* d_ws, size_t ws_size,
                              hipStream_t stream) {
    (void)in_sizes; (void)n_in; (void)out_size; (void)ws_size;
    const float* x      = (const float*)d_in[0];
    const float* w_qkv  = (const float*)d_in[1];
    const float* b_qkv  = (const float*)d_in[2];
    const float* w_proj = (const float*)d_in[3];
    const float* b_proj = (const float*)d_in[4];
    float* out = (float*)d_out;

    char* ws = (char*)d_ws;
    const size_t SZ_XB    = (size_t)ROWS * DMODEL * 2;               //  8 MB
    const size_t SZ_WQKV  = (size_t)QKVN * DMODEL * 2;               //  6 MB
    const size_t SZ_WPROJ = (size_t)DMODEL * DMODEL * 2;             //  2 MB
    const size_t SZ_HEADS = (size_t)BATCH * NHEADS * SEQ * HDIM * 2; //  8 MB

    bf16*  xb     = (bf16*)(ws);
    bf16*  wqkvT  = (bf16*)(ws + SZ_XB);
    bf16*  wprojT = (bf16*)(ws + SZ_XB + SZ_WQKV);
    bf16*  qbuf   = (bf16*)(ws + SZ_XB + SZ_WQKV + SZ_WPROJ);
    bf16*  kbuf   = (bf16*)((char*)qbuf + SZ_HEADS);
    bf16*  vbuf   = (bf16*)((char*)kbuf + SZ_HEADS);
    float* qkvf   = (float*)((char*)vbuf + SZ_HEADS);                // 48 MB
    bf16*  attnb  = (bf16*)qkvf;   // alias: qkv dead after RoPE + V transpose

    // 1) bf16 conversions / LDS-tiled weight transposes
    k_cvt_bf16_v4<<<(ROWS * DMODEL / 4 + 255) / 256, 256, 0, stream>>>(
        x, xb, ROWS * DMODEL / 4);
    k_transpose_bf16<<<(DMODEL / 32) * (QKVN / 32), 256, 0, stream>>>(
        w_qkv, wqkvT, DMODEL, QKVN);
    k_transpose_bf16<<<(DMODEL / 32) * (DMODEL / 32), 256, 0, stream>>>(
        w_proj, wprojT, DMODEL, DMODEL);

    // 2) QKV GEMM (fp32 out)
    k_gemm_bf16<<<(ROWS / 32) * (QKVN / 64) / 4, 128, 0, stream>>>(
        xb, wqkvT, b_qkv, qkvf, ROWS, QKVN, DMODEL);

    // 3) RoPE on Q/K + tiled V transpose
    k_rope_qk<<<(BATCH * NHEADS * SEQ * HDIM) / 256, 256, 0, stream>>>(
        qkvf, qbuf, kbuf);
    k_v_transpose<<<BATCH * NHEADS * 2 * (SEQ / 32), 256, 0, stream>>>(qkvf, vbuf);

    // 4) causal flash attention -> bf16 [4096 x 1024]
    k_flash_attn<<<(BATCH * NHEADS * (SEQ / 16)) / 4, 128, 0, stream>>>(
        qbuf, kbuf, vbuf, attnb);

    // 5) output projection -> fp32 d_out
    k_gemm_bf16<<<(ROWS / 32) * (DMODEL / 64) / 4, 128, 0, stream>>>(
        attnb, wprojT, b_proj, out, ROWS, DMODEL, DMODEL);
}
